// Attention_1511828488886
// MI455X (gfx1250) — compile-verified, matrix operand order checked by reading, not run on hardware
//
#include <hip/hip_runtime.h>
#include <hip/hip_bf16.h>
#include <math.h>

typedef __attribute__((ext_vector_type(16))) _Float16 v16h;
typedef __attribute__((ext_vector_type(8)))  _Float16 v8h;
typedef __attribute__((ext_vector_type(8)))  float    v8f;
typedef __attribute__((ext_vector_type(4)))  int      v4i;

#define BATCH   2
#define SEQ     2304
#define DIMC    256
#define HEADS   8
#define HDIM    32
#define TABLE_N 9025
#define CPBH    512
#define QKV_O   768

// Async global->LDS path (CDNA5 GLOBAL_LOAD_ASYNC_TO_LDS_B128 + ASYNCcnt),
// feature-detected so the file compiles on toolchains without the builtins.
#if defined(__has_builtin)
#if __has_builtin(__builtin_amdgcn_global_load_async_to_lds_b128) && \
    __has_builtin(__builtin_amdgcn_s_wait_asynccnt)
#define HAVE_ASYNC_LDS 1
#endif
#endif

__device__ __forceinline__ void stage_b128(const _Float16* g, _Float16* l) {
#ifdef HAVE_ASYNC_LDS
  __builtin_amdgcn_global_load_async_to_lds_b128(
      (__attribute__((address_space(1))) v4i*)g,
      (__attribute__((address_space(3))) v4i*)l, 0, 0);
#else
  *(v8h*)l = *(const v8h*)g;
#endif
}

__device__ __forceinline__ void stage_wait() {
#ifdef HAVE_ASYNC_LDS
  __builtin_amdgcn_s_wait_asynccnt(0);
#endif
}

// ---------------------------------------------------------------------------
// WMMA fragment helpers.
// 16-bit A-fragment (16x32, M x K), ISA layout: lane l holds row M = l%16;
// lanes 0-15: K = {0..7, 16..23}, lanes 16-31: K = {8..15, 24..31}.
// Caller passes p already offset by (half*8) halves; we read p[0..7], p[16..23].
// B-fragment (32x16, K x N) has the identical per-lane pattern with N = l%16,
// so any source contiguous-in-K for a fixed column uses the same load.
// ---------------------------------------------------------------------------
__device__ __forceinline__ v16h frag_from_f16(const _Float16* p) {
  v16h r;
#pragma unroll
  for (int i = 0; i < 8; ++i) { r[i] = p[i]; r[8 + i] = p[16 + i]; }
  return r;
}

__device__ __forceinline__ v16h frag_from_f32(const float* p) {
  v16h r;
#pragma unroll
  for (int i = 0; i < 8; ++i) {
    r[i]     = (_Float16)p[i];
    r[8 + i] = (_Float16)p[16 + i];
  }
  return r;
}

// ---------------------------------------------------------------------------
// CPB MLP: tab[row][h] = W2[h] . relu(coords[row] @ W1^T + b1) + b2[h]
// ---------------------------------------------------------------------------
__global__ void cpb_kernel(const float* __restrict__ coords,
                           const float* __restrict__ W1, const float* __restrict__ b1,
                           const float* __restrict__ W2, const float* __restrict__ b2,
                           float* __restrict__ tab) {
  int t = blockIdx.x * blockDim.x + threadIdx.x;
  if (t >= TABLE_N * HEADS) return;
  int h = t % HEADS, row = t / HEADS;
  float c0 = coords[row * 2], c1 = coords[row * 2 + 1];
  float acc = 0.f;
  for (int u = 0; u < CPBH; ++u) {
    float hid = fmaf(c0, W1[u * 2], fmaf(c1, W1[u * 2 + 1], b1[u]));
    hid = fmaxf(hid, 0.f);
    acc = fmaf(hid, W2[h * CPBH + u], acc);
  }
  tab[row * HEADS + h] = acc + b2[h];
}

// ---------------------------------------------------------------------------
// C[M x No] = A[M x K](f32) @ W[No x K]^T(f32) + bias, via f16 WMMA.
// One wave computes a 16x64 strip: A-fragment loaded once per K-chunk and
// reused for 4 WMMAs. K is a template constant -> fully unrolled (32 WMMAs).
// ---------------------------------------------------------------------------
template <int K>
__global__ __launch_bounds__(256) void gemm_f16wmma(
    const float* __restrict__ A, const float* __restrict__ W,
    const float* __restrict__ bias, float* __restrict__ C,
    int M, int No, int total_waves) {
  const int wid = blockIdx.x * (blockDim.x >> 5) + (threadIdx.x >> 5);
  if (wid >= total_waves) return;                 // wave-uniform exit
  const int lane = threadIdx.x & 31;
  const int l16 = lane & 15, half = lane >> 4;
  const int nstrips = No >> 6;                    // 64-wide strips
  const int tm = wid / nstrips, tn = wid % nstrips;

  const float* arow = A + (size_t)(tm * 16 + l16) * K + half * 8;
  const float* wbase = W + (size_t)(tn * 64 + l16) * K + half * 8;

  v8f acc[4] = {{}, {}, {}, {}};
#pragma unroll
  for (int c = 0; c < K; c += 32) {
    v16h af = frag_from_f32(arow + c);
#pragma unroll
    for (int t = 0; t < 4; ++t) {
      v16h bf = frag_from_f32(wbase + (size_t)t * 16 * K + c);
      acc[t] = __builtin_amdgcn_wmma_f32_16x16x32_f16(false, af, false, bf,
                                                      (short)0, acc[t], false, false);
    }
  }
#pragma unroll
  for (int t = 0; t < 4; ++t) {
    const int col = tn * 64 + t * 16 + l16;
    const float bv = bias[col];
#pragma unroll
    for (int r = 0; r < 8; ++r)
      C[(size_t)(tm * 16 + r + 8 * half) * No + col] = acc[t][r] + bv;
  }
}

// ---------------------------------------------------------------------------
// Prep: from qkv (B,N,768) f32 produce
//   qh (b,h,n,32) f16 = (l2norm(q) + query_emb[h]) * softplus(temp[h])*seq
//   kh (b,h,n,32) f16 =  l2norm(k)
//   vT (b,h,32,n) f16 =  v transposed (so P@V B-fragments are contiguous-in-K)
// ---------------------------------------------------------------------------
__global__ void prep_kernel(const float* __restrict__ qkv,
                            const float* __restrict__ temperature,
                            const float* __restrict__ seq_scale,
                            const float* __restrict__ qemb,
                            _Float16* __restrict__ qh,
                            _Float16* __restrict__ kh,
                            _Float16* __restrict__ vT) {
  int t = blockIdx.x * blockDim.x + threadIdx.x;
  if (t >= BATCH * SEQ * HEADS) return;
  int h = t % HEADS, n = (t / HEADS) % SEQ, b = t / (HEADS * SEQ);

  const float* row = qkv + (size_t)(b * SEQ + n) * QKV_O;
  const float* q = row + h * HDIM;
  const float* k = row + DIMC + h * HDIM;
  const float* v = row + 2 * DIMC + h * HDIM;

  float qs = 0.f, ks = 0.f;
#pragma unroll
  for (int d = 0; d < HDIM; ++d) { qs = fmaf(q[d], q[d], qs); ks = fmaf(k[d], k[d], ks); }
  float qn = 1.f / fmaxf(sqrtf(qs), 1e-12f);
  float kn = 1.f / fmaxf(sqrtf(ks), 1e-12f);
  float scale = log1pf(expf(temperature[h])) * seq_scale[0];

  size_t o  = ((size_t)(b * HEADS + h) * SEQ + n) * HDIM;
  size_t ot = ((size_t)(b * HEADS + h) * HDIM) * SEQ + n;
#pragma unroll
  for (int d = 0; d < HDIM; ++d) {
    qh[o + d] = (_Float16)((q[d] * qn + qemb[h * HDIM + d]) * scale);
    kh[o + d] = (_Float16)(k[d] * kn);
    vT[ot + (size_t)d * SEQ] = (_Float16)v[d];
  }
}

// ---------------------------------------------------------------------------
// Flash attention. Block = 4 waves sharing one (b,h); each wave owns 16 query
// rows. Per BN=64 kv tile:
//   - K-tile (64x32 f16) and V^T-tile (32x64 f16) staged cooperatively into
//     LDS (async-to-LDS when available), shared by all 4 waves.
//   - S tiles: v_wmma_f32_16x16x32_f16 (single K=32 = head_dim WMMA each)
//   - bias: gather tab[relidx[n,m]*8+h] (relidx 21.2 MB -> L2-resident)
//   - online softmax; row reductions via shfl_xor over 16-lane row groups
//   - O += P@V with P staged per-wave through LDS in A-fragment layout
// ---------------------------------------------------------------------------
#define ATT_WAVES 4
#define BN 64

__global__ __launch_bounds__(ATT_WAVES * 32) void attn_kernel(
    const _Float16* __restrict__ qh, const _Float16* __restrict__ kh,
    const _Float16* __restrict__ vT, const int* __restrict__ relidx,
    const float* __restrict__ tab, float* __restrict__ attn_out) {
  __shared__ __align__(16) _Float16 Kt[BN][HDIM];            // 4 KB
  __shared__ __align__(16) _Float16 Vt[HDIM][BN];            // 4 KB
  __shared__ __align__(16) _Float16 Pbuf[ATT_WAVES][16 * BN];// 8 KB

  const int tid  = threadIdx.x;
  const int wave = tid >> 5;
  const int lane = tid & 31;
  const int l16 = lane & 15, half = lane >> 4;

  const int strips_per_bh = SEQ / (16 * ATT_WAVES);
  const int bh    = blockIdx.x / strips_per_bh;
  const int strip = blockIdx.x % strips_per_bh;
  const int b = bh / HEADS, h = bh % HEADS;
  const int q0 = (strip * ATT_WAVES + wave) * 16;

  const size_t bh_off = (size_t)bh * SEQ * HDIM;
  const _Float16* qbase = qh + bh_off;
  const _Float16* kbase = kh + bh_off;
  const _Float16* vbase = vT + bh_off;

  // q strip A-fragment (16x32, K = head_dim), row stride 32 halves.
  const v16h aq = frag_from_f16(qbase + (size_t)(q0 + l16) * HDIM + half * 8);

  float m_run[8], l_run[8];
  v8f o0 = {}, o1 = {};
#pragma unroll
  for (int r = 0; r < 8; ++r) { m_run[r] = -3.0e38f; l_run[r] = 0.f; }

  _Float16* P = &Pbuf[wave][0];

  for (int kv0 = 0; kv0 < SEQ; kv0 += BN) {
    if (kv0 + BN < SEQ)
      __builtin_prefetch(kbase + (size_t)(kv0 + BN) * HDIM, 0, 1);

    // ---- cooperative K/V tile staging (128 threads x 2 b128 each/tile) ----
#pragma unroll
    for (int i = 0; i < 2; ++i) {
      const int c = tid + i * 128;                 // 0..255 b128 chunks
      const int krow = c >> 2, kpart = c & 3;      // K-tile: 64 rows x 4 chunks
      stage_b128(kbase + (size_t)(kv0 + krow) * HDIM + kpart * 8, &Kt[krow][kpart * 8]);
      const int vrow = c >> 3, vpart = c & 7;      // V-tile: 32 rows x 8 chunks
      stage_b128(vbase + (size_t)vrow * SEQ + kv0 + vpart * 8, &Vt[vrow][vpart * 8]);
    }
    stage_wait();
    __syncthreads();

    // ---- S = q @ k^T : 4 column tiles, single K=32 WMMA each --------------
    v8f s[4];
#pragma unroll
    for (int j = 0; j < 4; ++j) {
      v16h bk = frag_from_f16(&Kt[j * 16 + l16][0] + half * 8);
      v8f z = {};
      s[j] = __builtin_amdgcn_wmma_f32_16x16x32_f16(false, aq, false, bk,
                                                    (short)0, z, false, false);
    }
    // ---- + relative position bias (gather; idx coalesced across lanes) ----
#pragma unroll
    for (int j = 0; j < 4; ++j) {
      const int col = kv0 + j * 16 + l16;
#pragma unroll
      for (int r = 0; r < 8; ++r) {
        const int qr = q0 + r + 8 * half;
        const int idx = relidx[(size_t)qr * SEQ + col];
        s[j][r] += tab[idx * HEADS + h];
      }
    }
    // ---- online softmax ---------------------------------------------------
    float mt[8], alpha[8];
#pragma unroll
    for (int r = 0; r < 8; ++r) {
      float m = fmaxf(fmaxf(s[0][r], s[1][r]), fmaxf(s[2][r], s[3][r]));
#pragma unroll
      for (int off = 8; off >= 1; off >>= 1) m = fmaxf(m, __shfl_xor(m, off, 32));
      mt[r] = fmaxf(m_run[r], m);
      alpha[r] = __expf(m_run[r] - mt[r]);
    }
#pragma unroll
    for (int j = 0; j < 4; ++j)
#pragma unroll
      for (int r = 0; r < 8; ++r) s[j][r] = __expf(s[j][r] - mt[r]);
#pragma unroll
    for (int r = 0; r < 8; ++r) {
      float rs = (s[0][r] + s[1][r]) + (s[2][r] + s[3][r]);
#pragma unroll
      for (int off = 8; off >= 1; off >>= 1) rs += __shfl_xor(rs, off, 32);
      l_run[r] = l_run[r] * alpha[r] + rs;
      m_run[r] = mt[r];
      o0[r] *= alpha[r];
      o1[r] *= alpha[r];
    }
    // ---- P -> LDS (row-major 16 x BN f16), consumed as A-fragments --------
#pragma unroll
    for (int j = 0; j < 4; ++j)
#pragma unroll
      for (int r = 0; r < 8; ++r)
        P[(r + 8 * half) * BN + j * 16 + l16] = (_Float16)s[j][r];
    asm volatile("s_wait_dscnt 0x0" ::: "memory");

    // ---- O += P @ V : 2 K-chunks x 2 d-tiles ------------------------------
#pragma unroll
    for (int kc = 0; kc < 2; ++kc) {
      v16h ap = frag_from_f16(P + l16 * BN + kc * 32 + half * 8);
#pragma unroll
      for (int dt = 0; dt < 2; ++dt) {
        v16h bv = frag_from_f16(&Vt[dt * 16 + l16][kc * 32] + half * 8);
        if (dt == 0)
          o0 = __builtin_amdgcn_wmma_f32_16x16x32_f16(false, ap, false, bv,
                                                      (short)0, o0, false, false);
        else
          o1 = __builtin_amdgcn_wmma_f32_16x16x32_f16(false, ap, false, bv,
                                                      (short)0, o1, false, false);
      }
    }
    __syncthreads();   // all waves done with Kt/Vt before next staging
  }

  // ---- normalize, write (b, n, h*32 + d) for the projection GEMM ----------
#pragma unroll
  for (int r = 0; r < 8; ++r) {
    const float inv = 1.f / l_run[r];
    const int n = q0 + r + 8 * half;
    float* orow = attn_out + (size_t)(b * SEQ + n) * DIMC + h * HDIM;
    orow[l16]      = o0[r] * inv;
    orow[16 + l16] = o1[r] * inv;
  }
}

// ---------------------------------------------------------------------------
extern "C" void kernel_launch(void* const* d_in, const int* in_sizes, int n_in,
                              void* d_out, int out_size, void* d_ws, size_t ws_size,
                              hipStream_t stream) {
  (void)in_sizes; (void)n_in; (void)out_size; (void)ws_size;

  const float* x      = (const float*)d_in[0];
  const int*   relidx = (const int*)d_in[1];
  const float* coords = (const float*)d_in[2];
  const float* seqsc  = (const float*)d_in[3];
  // d_in[4] = padding_mask (unused by the reference)
  const float* Wqkv   = (const float*)d_in[5];
  const float* bqkv   = (const float*)d_in[6];
  const float* temp   = (const float*)d_in[7];
  const float* qemb   = (const float*)d_in[8];
  const float* Wproj  = (const float*)d_in[9];
  const float* bproj  = (const float*)d_in[10];
  const float* Wc1    = (const float*)d_in[11];
  const float* bc1    = (const float*)d_in[12];
  const float* Wc2    = (const float*)d_in[13];
  const float* bc2    = (const float*)d_in[14];

  char* ws = (char*)d_ws;
  size_t off = 0;
  auto wsa = [&](size_t bytes) -> char* {
    char* p = ws + off;
    off = (off + bytes + 255) & ~(size_t)255;
    return p;
  };
  float*    qkv      = (float*)   wsa((size_t)BATCH * SEQ * QKV_O * 4);       // 14.2 MB
  float*    tab      = (float*)   wsa((size_t)TABLE_N * HEADS * 4);           // 0.3 MB
  _Float16* qh       = (_Float16*)wsa((size_t)BATCH * HEADS * SEQ * HDIM * 2);// 2.4 MB
  _Float16* kh       = (_Float16*)wsa((size_t)BATCH * HEADS * SEQ * HDIM * 2);// 2.4 MB
  _Float16* vT       = (_Float16*)wsa((size_t)BATCH * HEADS * HDIM * SEQ * 2);// 2.4 MB
  float*    attn_out = (float*)   wsa((size_t)BATCH * SEQ * DIMC * 4);        // 4.7 MB

  // 1) CPB MLP table
  cpb_kernel<<<(TABLE_N * HEADS + 255) / 256, 256, 0, stream>>>(
      coords, Wc1, bc1, Wc2, bc2, tab);

  // 2) QKV projection: (4608 x 256) @ (256 x 768)
  {
    const int M = BATCH * SEQ, No = QKV_O;
    const int waves = (M / 16) * (No / 64);
    gemm_f16wmma<DIMC><<<(waves + 7) / 8, 256, 0, stream>>>(x, Wqkv, bqkv, qkv,
                                                            M, No, waves);
  }

  // 3) l2norm / scale / f16 convert / V transpose
  prep_kernel<<<(BATCH * SEQ * HEADS + 255) / 256, 256, 0, stream>>>(
      qkv, temp, seqsc, qemb, qh, kh, vT);

  // 4) flash attention with fused rel-bias gather
  attn_kernel<<<BATCH * HEADS * (SEQ / (16 * ATT_WAVES)), ATT_WAVES * 32, 0, stream>>>(
      qh, kh, vT, relidx, tab, attn_out);

  // 5) output projection: (4608 x 256) @ (256 x 256) -> d_out
  {
    const int M = BATCH * SEQ, No = DIMC;
    const int waves = (M / 16) * (No / 64);
    gemm_f16wmma<DIMC><<<(waves + 7) / 8, 256, 0, stream>>>(attn_out, Wproj, bproj,
                                                            (float*)d_out, M, No, waves);
  }
}
